// GLHAD_16269336117633
// MI455X (gfx1250) — matrix operand order repaired and checked by reading, not compiled, not run on hardware
//
#include <hip/hip_runtime.h>
#include <hip/hip_bf16.h>
#include <math.h>

#define NN   50000
#define EE   800000
#define CC   128
#define HH   128
#define OUTD 2

typedef __attribute__((ext_vector_type(16))) __bf16 v16bf;
typedef __attribute__((ext_vector_type(2)))  __bf16 bf2;
typedef __attribute__((ext_vector_type(8)))  float  v8f;

// Native RNE fptrunc: lowers to hardware v_cvt_*bf16_f32 on gfx1250 —
// far fewer VALU ops than a manual round-to-nearest-even bit sequence.
__device__ __forceinline__ __bf16 f2bf(float f) { return (__bf16)f; }

__device__ __forceinline__ bf2 pk2(float a, float b) {
  bf2 r; r.x = (__bf16)a; r.y = (__bf16)b; return r;
}

// A-fragment assembled as 8 packed bf16 pairs (dword-granular writes, no
// half-register merges).
union AFrag { v16bf v; bf2 h[8]; };

__device__ __forceinline__ void load_run8(AFrag& f, int pairBase, const float* p) {
  float4 q0 = ((const float4*)p)[0];
  float4 q1 = ((const float4*)p)[1];
  f.h[pairBase + 0] = pk2(q0.x, q0.y);
  f.h[pairBase + 1] = pk2(q0.z, q0.w);
  f.h[pairBase + 2] = pk2(q1.x, q1.y);
  f.h[pairBase + 3] = pk2(q1.z, q1.w);
}

// ---------------------------------------------------------------------------
// Kernel 1: per-edge MLP  h = silu([x_dst,x_src]@W1+b1); p = sigmoid(h@w2+b2)
// WMMA GEMM: M=32 edges/wave (two A tiles sharing each B fragment),
// K=256, N=128.  Scatters local_homo numerator + degree.
// ---------------------------------------------------------------------------
__global__ __launch_bounds__(256) void edge_mlp_kernel(
    const float* __restrict__ x, const int* __restrict__ ei,
    const float* __restrict__ w1, const float* __restrict__ b1,
    const float* __restrict__ w2, const float* __restrict__ b2,
    float* __restrict__ pe_out, float* __restrict__ lh_num, float* __restrict__ deg)
{
  __shared__ __bf16 sW[2 * CC * HH];          // 64 KB: sW[n*256 + k] = w1[k][n]
  const int t = threadIdx.x;
  for (int idx = t; idx < 2 * CC * HH; idx += 256) {
    int k = idx >> 7;          // 0..255 (rows of w1)
    int n = idx & 127;         // 0..127
    sW[n * 256 + k] = f2bf(w1[idx]);          // coalesced global read
  }
  __syncthreads();

  const int wave = t >> 5;
  const int lane = t & 31;
  const int h    = lane >> 4;                 // K-half selector
  const int ln   = lane & 15;
  const int tb   = blockIdx.x * 256 + wave * 32;   // E = 800000 divides by 256

  // Two edge tiles per wave (A layout: lane%16 == M row)
  const int e0 = tb + ln;
  const int e1 = tb + 16 + ln;
  const int s0 = ei[e0],      s1 = ei[e1];
  const int d0 = ei[EE + e0], d1 = ei[EE + e1];
  const float* rowD0 = x + (size_t)d0 * CC;   // feature k<128  -> x[dst]
  const float* rowS0 = x + (size_t)s0 * CC;   // feature k>=128 -> x[src]
  const float* rowD1 = x + (size_t)d1 * CC;
  const float* rowS1 = x + (size_t)s1 * CC;

  v8f acc0[8], acc1[8];
  #pragma unroll
  for (int nn = 0; nn < 8; ++nn) {
    acc0[nn] = v8f{0.f,0.f,0.f,0.f,0.f,0.f,0.f,0.f};
    acc1[nn] = v8f{0.f,0.f,0.f,0.f,0.f,0.f,0.f,0.f};
  }

  #pragma unroll
  for (int kk = 0; kk < 8; ++kk) {            // K = 256 = 8 * 32
    const int k0 = kk * 32 + h * 16;          // run 0: k0..k0+7
    const int k1 = k0 + 8;                    // run 1: k1..k1+7
    AFrag a0, a1;
    load_run8(a0, 0, (k0 < CC) ? (rowD0 + k0) : (rowS0 + (k0 - CC)));
    load_run8(a0, 4, (k1 < CC) ? (rowD0 + k1) : (rowS0 + (k1 - CC)));
    load_run8(a1, 0, (k0 < CC) ? (rowD1 + k0) : (rowS1 + (k0 - CC)));
    load_run8(a1, 4, (k1 < CC) ? (rowD1 + k1) : (rowS1 + (k1 - CC)));
    #pragma unroll
    for (int nn = 0; nn < 8; ++nn) {          // N = 128 = 8 * 16
      const v16bf b = *(const v16bf*)(sW + (nn * 16 + ln) * 256 + kk * 32 + h * 16);
      acc0[nn] = __builtin_amdgcn_wmma_f32_16x16x32_bf16(
          false, a0.v, false, b, (short)0, acc0[nn], false, false);
      acc1[nn] = __builtin_amdgcn_wmma_f32_16x16x32_bf16(
          false, a1.v, false, b, (short)0, acc1[nn], false, false);
    }
  }

  // Epilogue: silu(+b1), dot with w2, reduce across 16-lane half-groups.
  float b1v[8], w2v[8];
  #pragma unroll
  for (int nn = 0; nn < 8; ++nn) { int n = nn * 16 + ln; b1v[nn] = b1[n]; w2v[nn] = w2[n]; }
  const float bb2 = b2[0];

  #pragma unroll
  for (int tile = 0; tile < 2; ++tile) {
    const v8f* acc = tile ? acc1 : acc0;
    const int  eb  = tb + tile * 16;
    float part[8];
    #pragma unroll
    for (int v = 0; v < 8; ++v) {             // row m = h*8 + v
      float s = 0.f;
      #pragma unroll
      for (int nn = 0; nn < 8; ++nn) {
        float pre = acc[nn][v] + b1v[nn];
        float sil = pre / (1.f + __expf(-pre));
        s += sil * w2v[nn];
      }
      #pragma unroll
      for (int m = 1; m < 16; m <<= 1) s += __shfl_xor(s, m, 32);
      part[v] = s;
    }
    if (ln == 0) {
      #pragma unroll
      for (int v = 0; v < 8; ++v) {
        int e2 = eb + h * 8 + v;
        float z = part[v] + bb2;
        float p = 1.f / (1.f + __expf(-z));
        pe_out[e2] = p;
        int d = ei[EE + e2];
        atomicAdd(lh_num + d, p);
        atomicAdd(deg + d, 1.0f);
      }
    }
  }
}

// ---------------------------------------------------------------------------
__global__ void lh_finalize(const float* __restrict__ num,
                            const float* __restrict__ deg,
                            float* __restrict__ lh_out)
{
  int i = blockIdx.x * 256 + threadIdx.x;
  if (i < NN) lh_out[i] = num[i] / fmaxf(deg[i], 1.0f);
}

// ---------------------------------------------------------------------------
// SPMM scatter: aggL[dst] += fl[e]*xt[src]; aggH[dst] += fh[e]*xt[src]
// 32 lanes per edge, float4 per lane (L2-resident traffic).
// ---------------------------------------------------------------------------
__global__ __launch_bounds__(256) void spmm2_kernel(
    const float* __restrict__ xt, const int* __restrict__ ei,
    const float* __restrict__ fl, const float* __restrict__ fh,
    float* __restrict__ aggL, float* __restrict__ aggH)
{
  long long gid = (long long)blockIdx.x * 256 + threadIdx.x;
  int e = (int)(gid >> 5);
  int c = ((int)gid & 31) * 4;
  if (e >= EE) return;
  int s = ei[e], d = ei[EE + e];
  float wl = fl[e], wh = fh[e];
  const float4 v = *(const float4*)(xt + (size_t)s * CC + c);
  float* pL = aggL + (size_t)d * CC + c;
  float* pH = aggH + (size_t)d * CC + c;
  atomicAdd(pL + 0, wl * v.x); atomicAdd(pL + 1, wl * v.y);
  atomicAdd(pL + 2, wl * v.z); atomicAdd(pL + 3, wl * v.w);
  atomicAdd(pH + 0, wh * v.x); atomicAdd(pH + 1, wh * v.y);
  atomicAdd(pH + 2, wh * v.z); atomicAdd(pH + 3, wh * v.w);
}

// ---------------------------------------------------------------------------
// Conv: xL = relu(l2norm(aggL@lw)); xH = relu(l2norm(aggH@hw));
//       xt_out = lh*xL + (1-lh)*xH          (fused dual WMMA GEMM)
// ---------------------------------------------------------------------------
__global__ __launch_bounds__(256) void conv_kernel(
    const float* __restrict__ aggL, const float* __restrict__ aggH,
    const float* __restrict__ lw,   const float* __restrict__ hw,
    const float* __restrict__ lh,   float* __restrict__ xt_out)
{
  __shared__ __bf16 sLW[CC * HH];             // 32 KB each, transposed
  __shared__ __bf16 sHW[CC * HH];
  const int t = threadIdx.x;
  for (int idx = t; idx < CC * HH; idx += 256) {
    int k = idx >> 7, n = idx & 127;
    sLW[n * 128 + k] = f2bf(lw[idx]);
    sHW[n * 128 + k] = f2bf(hw[idx]);
  }
  __syncthreads();

  const int wave = t >> 5, lane = t & 31, h = lane >> 4, ln = lane & 15;
  const int tb = blockIdx.x * 128 + wave * 16;
  const int r  = tb + ln;
  const int rc = (r < NN) ? r : (NN - 1);     // clamp keeps EXEC uniform for WMMA
  const float* rL = aggL + (size_t)rc * CC;
  const float* rH = aggH + (size_t)rc * CC;

  v8f accL[8], accH[8];
  #pragma unroll
  for (int nn = 0; nn < 8; ++nn) {
    accL[nn] = v8f{0.f,0.f,0.f,0.f,0.f,0.f,0.f,0.f};
    accH[nn] = v8f{0.f,0.f,0.f,0.f,0.f,0.f,0.f,0.f};
  }

  #pragma unroll
  for (int kk = 0; kk < 4; ++kk) {            // K = 128 = 4 * 32
    const int k0 = kk * 32 + h * 16;
    const int k1 = k0 + 8;
    AFrag aL, aH;
    load_run8(aL, 0, rL + k0);
    load_run8(aL, 4, rL + k1);
    load_run8(aH, 0, rH + k0);
    load_run8(aH, 4, rH + k1);
    #pragma unroll
    for (int nn = 0; nn < 8; ++nn) {
      const v16bf bL = *(const v16bf*)(sLW + (nn * 16 + ln) * 128 + kk * 32 + h * 16);
      const v16bf bH = *(const v16bf*)(sHW + (nn * 16 + ln) * 128 + kk * 32 + h * 16);
      accL[nn] = __builtin_amdgcn_wmma_f32_16x16x32_bf16(
          false, aL.v, false, bL, (short)0, accL[nn], false, false);
      accH[nn] = __builtin_amdgcn_wmma_f32_16x16x32_bf16(
          false, aH.v, false, bH, (short)0, accH[nn], false, false);
    }
  }

  #pragma unroll
  for (int v = 0; v < 8; ++v) {               // row m = h*8 + v
    const int row = tb + h * 8 + v;
    float ssL = 0.f, ssH = 0.f;
    #pragma unroll
    for (int nn = 0; nn < 8; ++nn) {
      ssL += accL[nn][v] * accL[nn][v];
      ssH += accH[nn][v] * accH[nn][v];
    }
    #pragma unroll
    for (int m = 1; m < 16; m <<= 1) {
      ssL += __shfl_xor(ssL, m, 32);
      ssH += __shfl_xor(ssH, m, 32);
    }
    const float invL = 1.f / fmaxf(sqrtf(ssL), 1e-12f);
    const float invH = 1.f / fmaxf(sqrtf(ssH), 1e-12f);
    if (row < NN) {
      const float g = lh[row];
      #pragma unroll
      for (int nn = 0; nn < 8; ++nn) {
        float vl = fmaxf(accL[nn][v] * invL, 0.f);
        float vh = fmaxf(accH[nn][v] * invH, 0.f);
        xt_out[(size_t)row * CC + nn * 16 + ln] = g * vl + (1.f - g) * vh;
      }
    }
  }
}

// ---------------------------------------------------------------------------
__global__ void out_kernel(const float* __restrict__ xt, const float* __restrict__ W,
                           const float* __restrict__ b, float* __restrict__ out)
{
  int i = blockIdx.x * 256 + threadIdx.x;
  if (i >= NN) return;
  const float* row = xt + (size_t)i * CC;
  float ss = 0.f, a0 = 0.f, a1 = 0.f;
  #pragma unroll 4
  for (int k = 0; k < CC; ++k) {
    float v = row[k];
    ss += v * v;
    a0 += v * W[k * 2 + 0];
    a1 += v * W[k * 2 + 1];
  }
  float inv = 1.f / fmaxf(sqrtf(ss), 1e-12f);
  out[i * 2 + 0] = a0 * inv + b[0];
  out[i * 2 + 1] = a1 * inv + b[1];
}

// ---------------------------------------------------------------------------
extern "C" void kernel_launch(void* const* d_in, const int* in_sizes, int n_in,
                              void* d_out, int out_size, void* d_ws, size_t ws_size,
                              hipStream_t stream)
{
  (void)in_sizes; (void)n_in; (void)out_size; (void)ws_size;
  const float* x   = (const float*)d_in[0];
  const int*   ei  = (const int*)  d_in[1];
  const float* fl  = (const float*)d_in[2];
  const float* fh  = (const float*)d_in[3];
  const float* w1  = (const float*)d_in[4];
  const float* b1  = (const float*)d_in[5];
  const float* w2  = (const float*)d_in[6];
  const float* b2  = (const float*)d_in[7];
  const float* lW1 = (const float*)d_in[8];
  const float* hW1 = (const float*)d_in[9];
  const float* lW2 = (const float*)d_in[10];
  const float* hW2 = (const float*)d_in[11];
  const float* oW  = (const float*)d_in[12];
  const float* ob  = (const float*)d_in[13];

  float* out    = (float*)d_out;
  float* lh_out = out + (size_t)NN * OUTD;          // local_homo output slot
  float* pe_out = lh_out + NN;                      // pre_edge_type output slot

  float* ws     = (float*)d_ws;
  float* lh_num = ws;
  float* deg    = ws + NN;
  float* aggL   = ws + 2 * (size_t)NN;
  float* aggH   = aggL + (size_t)NN * CC;
  float* xt1    = aggH + (size_t)NN * CC;

  // local homophily (edge MLP GEMM via WMMA)
  hipMemsetAsync(lh_num, 0, 2 * (size_t)NN * sizeof(float), stream);
  edge_mlp_kernel<<<EE / 256, 256, 0, stream>>>(x, ei, w1, b1, w2, b2,
                                                pe_out, lh_num, deg);
  lh_finalize<<<(NN + 255) / 256, 256, 0, stream>>>(lh_num, deg, lh_out);

  const int spmmGrid = (EE * 32) / 256;
  const int convGrid = (NN + 127) / 128;

  // conv layer 1 (input = x)
  hipMemsetAsync(aggL, 0, 2 * (size_t)NN * CC * sizeof(float), stream);
  spmm2_kernel<<<spmmGrid, 256, 0, stream>>>(x, ei, fl, fh, aggL, aggH);
  conv_kernel<<<convGrid, 256, 0, stream>>>(aggL, aggH, lW1, hW1, lh_out, xt1);

  // conv layer 2 (input = xt1); result overwrites aggL (each wave reads its
  // rows fully into registers before storing; clamped cross-reads are discarded)
  hipMemsetAsync(aggL, 0, 2 * (size_t)NN * CC * sizeof(float), stream);
  spmm2_kernel<<<spmmGrid, 256, 0, stream>>>(xt1, ei, fl, fh, aggL, aggH);
  conv_kernel<<<convGrid, 256, 0, stream>>>(aggL, aggH, lW2, hW2, lh_out, aggL);

  // output head
  out_kernel<<<(NN + 255) / 256, 256, 0, stream>>>(aggL, oW, ob, out);
}
